// myModel_DBLP_vae_77953656422622
// MI455X (gfx1250) — matrix operand order repaired and checked by reading, not compiled, not run on hardware
//
#include <hip/hip_runtime.h>
#include <math.h>

// ---------------------------------------------------------------------------
// Types for CDNA5 WMMA (wave32): D(16x16 f32) = A(16x32 bf16) * B(32x16 bf16) + C
// ---------------------------------------------------------------------------
typedef __attribute__((ext_vector_type(16))) __bf16       v16bf;
typedef __attribute__((ext_vector_type(8)))  float        v8f;
typedef __attribute__((ext_vector_type(4)))  float        f32x4;
typedef __attribute__((ext_vector_type(4)))  unsigned int u32x4;
typedef __attribute__((ext_vector_type(8)))  unsigned int u32x8;

static constexpr int BM = 128;     // block tile M
static constexpr int BN = 128;     // block tile N
static constexpr int BK = 64;      // block tile K (2 wmma k-steps of 32)
static constexpr int LDP = BK + 8; // LDS row pitch in bf16 elems (144B: breaks bank conflicts, keeps 16B align)
static constexpr int NTHREADS = 256; // 8 waves of 32

// Pack two f32 -> two bf16 (round-to-nearest via +0x8000 bias) in 3 VALU: add,add,v_perm_b32
__device__ __forceinline__ unsigned int pack_bf16x2(float f0, float f1) {
  unsigned int u0 = __builtin_bit_cast(unsigned int, f0) + 0x8000u;
  unsigned int u1 = __builtin_bit_cast(unsigned int, f1) + 0x8000u;
  // D bytes: [0]=f0.b2 [1]=f0.b3 [2]=f1.b2 [3]=f1.b3  (S0=u1 -> codes 4..7, S1=u0 -> codes 0..3)
  return __builtin_amdgcn_perm(u1, u0, 0x07060302u);
}
__device__ __forceinline__ unsigned short f2bf1(float f) {
  unsigned int u = __builtin_bit_cast(unsigned int, f) + 0x8000u;
  return (unsigned short)(u >> 16);
}

// NaN-safe fast tanh on v_exp_f32
__device__ __forceinline__ float fast_tanh(float x) {
  float e = __expf(2.0f * fabsf(x));
  float t = 1.0f - 2.0f / (e + 1.0f);
  return copysignf(t, x);
}

// Two 16B ds_load_b128 -> one 32B bf16x16 WMMA fragment
__device__ __forceinline__ v16bf ld_frag(const unsigned short* p0, const unsigned short* p1) {
  u32x4 lo = *reinterpret_cast<const u32x4*>(p0);
  u32x4 hi = *reinterpret_cast<const u32x4*>(p1);
  u32x8 w;
  w[0] = lo[0]; w[1] = lo[1]; w[2] = lo[2]; w[3] = lo[3];
  w[4] = hi[0]; w[5] = hi[1]; w[6] = hi[2]; w[7] = hi[3];
  return __builtin_bit_cast(v16bf, w);
}

// ---------------------------------------------------------------------------
// C[M,N] = act( op(A) @ op(B) + bias ),  f32 in / f32 out, bf16 WMMA inside.
//   TRANS_A == 1 : logical A[m,k] = Aptr[k*lda + m]  (pooling H^T @ lat; NT loads)
//   B_KN   == 1  : B given as [K,N] row-major, else weights W[N,K] (C = A @ W^T)
//   ACT: 0 none, 1 tanh, 2 relu
// Register-staged double-buffered LDS pipeline: global loads for tile t+2 are
// issued while tile t is computed; conversion+ds_store fill the alternate buffer.
// ---------------------------------------------------------------------------
template<int TRANS_A, int B_KN, int ACT>
__global__ __launch_bounds__(NTHREADS) void gemm_bf16_wmma(
    const float* __restrict__ A, const float* __restrict__ B,
    const float* __restrict__ bias, float* __restrict__ C,
    int M, int N, int K, int lda, int ldb, int ldc)
{
  __shared__ __align__(16) unsigned short sA[2][BM * LDP];
  __shared__ __align__(16) unsigned short sB[2][BN * LDP];

  const int m0   = blockIdx.y * BM;
  const int n0   = blockIdx.x * BN;
  const int tid  = threadIdx.x;
  const int wave = tid >> 5;
  const int lane = tid & 31;
  const int wm   = (wave & 1) * 64;  // 2 waves down M
  const int wn   = (wave >> 1) * 32; // 4 waves across N
  const int lr   = lane & 15;
  const int lh   = lane >> 4;

  f32x4 ra[8], rb[8];                // register staging for one K-tile of A and B

  auto load_tileA = [&](int k0) {
#pragma unroll
    for (int i = 0; i < 8; ++i) {
      int flat = tid + i * NTHREADS;
      if constexpr (!TRANS_A) {
        int r = flat >> 4, c4 = (flat & 15) << 2;        // [BM rows][16 float4 along K]
        ra[i] = *reinterpret_cast<const f32x4*>(A + (size_t)(m0 + r) * lda + (k0 + c4));
      } else {
        int r = flat >> 5, c4 = (flat & 31) << 2;        // [BK rows(k)][32 float4 along m]
        ra[i] = __builtin_nontemporal_load(
            reinterpret_cast<const f32x4*>(A + (size_t)(k0 + r) * lda + (m0 + c4)));
      }
    }
  };
  auto load_tileB = [&](int k0) {
#pragma unroll
    for (int i = 0; i < 8; ++i) {
      int flat = tid + i * NTHREADS;
      if constexpr (!B_KN) {
        int r = flat >> 4, c4 = (flat & 15) << 2;        // W[N,K]
        rb[i] = *reinterpret_cast<const f32x4*>(B + (size_t)(n0 + r) * ldb + (k0 + c4));
      } else {
        int r = flat >> 5, c4 = (flat & 31) << 2;        // B[K,N]
        rb[i] = *reinterpret_cast<const f32x4*>(B + (size_t)(k0 + r) * ldb + (n0 + c4));
      }
    }
  };
  auto store_tiles = [&](int buf) {
#pragma unroll
    for (int i = 0; i < 8; ++i) {
      int flat = tid + i * NTHREADS;
      if constexpr (!TRANS_A) {
        int r = flat >> 4, c4 = (flat & 15) << 2;
        *reinterpret_cast<uint2*>(&sA[buf][r * LDP + c4]) =
            make_uint2(pack_bf16x2(ra[i][0], ra[i][1]), pack_bf16x2(ra[i][2], ra[i][3]));
      } else {
        int r = flat >> 5, c4 = (flat & 31) << 2;        // transpose into sA[m][k]
        sA[buf][(c4 + 0) * LDP + r] = f2bf1(ra[i][0]);
        sA[buf][(c4 + 1) * LDP + r] = f2bf1(ra[i][1]);
        sA[buf][(c4 + 2) * LDP + r] = f2bf1(ra[i][2]);
        sA[buf][(c4 + 3) * LDP + r] = f2bf1(ra[i][3]);
      }
    }
#pragma unroll
    for (int i = 0; i < 8; ++i) {
      int flat = tid + i * NTHREADS;
      if constexpr (!B_KN) {
        int r = flat >> 4, c4 = (flat & 15) << 2;
        *reinterpret_cast<uint2*>(&sB[buf][r * LDP + c4]) =
            make_uint2(pack_bf16x2(rb[i][0], rb[i][1]), pack_bf16x2(rb[i][2], rb[i][3]));
      } else {
        int r = flat >> 5, c4 = (flat & 31) << 2;        // transpose into sB[n][k]
        sB[buf][(c4 + 0) * LDP + r] = f2bf1(rb[i][0]);
        sB[buf][(c4 + 1) * LDP + r] = f2bf1(rb[i][1]);
        sB[buf][(c4 + 2) * LDP + r] = f2bf1(rb[i][2]);
        sB[buf][(c4 + 3) * LDP + r] = f2bf1(rb[i][3]);
      }
    }
  };

  v8f acc[4][2];
#pragma unroll
  for (int i = 0; i < 4; ++i)
#pragma unroll
    for (int j = 0; j < 2; ++j)
#pragma unroll
      for (int v = 0; v < 8; ++v) acc[i][j][v] = 0.0f;

  const int ktiles = K / BK;

  // ---- pipeline prologue ----
  load_tileA(0); load_tileB(0);
  store_tiles(0);
  if (ktiles > 1) { load_tileA(BK); load_tileB(BK); }   // in flight across tile-0 compute
  __syncthreads();

  for (int kt = 0; kt < ktiles; ++kt) {
    const int buf = kt & 1;
    const unsigned short* pA = sA[buf];
    const unsigned short* pB = sB[buf];

    // ---- two wmma k-steps of 32 over the LDS tile ----
#pragma unroll
    for (int ks = 0; ks < 2; ++ks) {
      const int kb = ks * 32;
      v16bf af[4];
#pragma unroll
      for (int im = 0; im < 4; ++im) {
        // ISA A layout (16-bit 16x32): lanes 0-15 K=[0..7],[16..23]; lanes 16-31 K=[8..15],[24..31]
        const int base = (wm + im * 16 + lr) * LDP + kb + lh * 8;
        af[im] = ld_frag(&pA[base], &pA[base + 16]);
      }
      v16bf bfr[2];
#pragma unroll
      for (int in = 0; in < 2; ++in) {
        // B layout (16-bit 32x16), matching A: lanes 0-15 K=[0..7],[16..23]; lanes 16-31 K=[8..15],[24..31]; N = lane&15
        const int base = (wn + in * 16 + lr) * LDP + kb + lh * 8;
        bfr[in] = ld_frag(&pB[base], &pB[base + 16]);
      }
#pragma unroll
      for (int im = 0; im < 4; ++im)
#pragma unroll
        for (int in = 0; in < 2; ++in)
          acc[im][in] = __builtin_amdgcn_wmma_f32_16x16x32_bf16(
              false, af[im], false, bfr[in], (short)0, acc[im][in], false, false);
    }

    // ---- refill the other buffer; kick off loads for tile kt+2 ----
    if (kt + 1 < ktiles) {
      __syncthreads();                       // everyone done reading lds[buf^1] (iter kt-1)
      store_tiles(buf ^ 1);                  // consumes staged regs (tile kt+1)
      if (kt + 2 < ktiles) { load_tileA((kt + 2) * BK); load_tileB((kt + 2) * BK); }
      __syncthreads();                       // lds[buf^1] ready
    }
  }

  // ---- epilogue: + bias, activation, f32 store ----
#pragma unroll
  for (int im = 0; im < 4; ++im) {
#pragma unroll
    for (int in = 0; in < 2; ++in) {
      const int n_g = n0 + wn + in * 16 + lr;
      const float bv = bias ? bias[n_g] : 0.0f;
      const int mb = m0 + wm + im * 16 + lh * 8;  // C/D layout: VGPR v -> M = v + 8*(lane>=16)
#pragma unroll
      for (int v = 0; v < 8; ++v) {
        float r = acc[im][in][v] + bv;
        if constexpr (ACT == 1) r = fast_tanh(r);
        if constexpr (ACT == 2) r = fmaxf(r, 0.0f);
        C[(size_t)(mb + v) * ldc + n_g] = r;
      }
    }
  }
}

// ---------------------------------------------------------------------------
// lat = tanh( eps * exp(s) + m ),  eps ~ N(0,1) via counter-based hash + Box-Muller
// (JAX threefry bits are not reproducible here; deterministic per (seed,idx))
// ---------------------------------------------------------------------------
__device__ __forceinline__ unsigned int mixu(unsigned int x) {
  x ^= x >> 16; x *= 0x7feb352du;
  x ^= x >> 15; x *= 0x846ca68bu;
  x ^= x >> 16;
  return x;
}

__global__ void reparam_kernel(const float* __restrict__ m, const float* __restrict__ s,
                               float* __restrict__ lat, int n, unsigned int seed) {
  int i = blockIdx.x * blockDim.x + threadIdx.x;
  if (i >= n) return;
  unsigned int a = mixu((unsigned int)i * 0x9E3779B9u + seed);
  unsigned int b = mixu(a ^ 0x85EBCA6Bu);
  float u1 = (float)(a >> 8) * (1.0f / 16777216.0f) + 1e-7f;
  float u2 = (float)(b >> 8) * (1.0f / 16777216.0f);
  float eps = sqrtf(-2.0f * __logf(u1)) * __cosf(6.28318530718f * u2);
  lat[i] = fast_tanh(eps * __expf(s[i]) + m[i]);
}

// emb[:, 0:128] = latP ; emb[:, 128:256] = sum
__global__ void concat_kernel(const float* __restrict__ latP, const float* __restrict__ sum,
                              float* __restrict__ emb, int n /* = E*256 */) {
  int i = blockIdx.x * blockDim.x + threadIdx.x;
  if (i >= n) return;
  int row = i >> 8, c = i & 255;
  emb[i] = (c < 128) ? latP[row * 128 + c] : sum[row * 128 + (c - 128)];
}

// ---------------------------------------------------------------------------
extern "C" void kernel_launch(void* const* d_in, const int* in_sizes, int n_in,
                              void* d_out, int out_size, void* d_ws, size_t ws_size,
                              hipStream_t stream) {
  (void)in_sizes; (void)n_in; (void)out_size; (void)ws_size;
  constexpr int E = 8192, H1 = 512, H2 = 128, H3 = 256, DP = 1024;

  const float* H_[4] = { (const float*)d_in[0], (const float*)d_in[1],
                         (const float*)d_in[2], (const float*)d_in[3] };
  struct Branch { const float *W1, *b1, *Wm, *bm, *Ws, *bs; };
  Branch br[4];
  for (int b = 0; b < 4; ++b) {
    int o = 4 + b * 6;
    br[b] = { (const float*)d_in[o+0], (const float*)d_in[o+1], (const float*)d_in[o+2],
              (const float*)d_in[o+3], (const float*)d_in[o+4], (const float*)d_in[o+5] };
  }
  struct Ev { const float *W1, *b1, *W2, *b2; };
  Ev ev[3];
  for (int e = 0; e < 3; ++e) {
    int o = 28 + e * 4;
    ev[e] = { (const float*)d_in[o+0], (const float*)d_in[o+1],
              (const float*)d_in[o+2], (const float*)d_in[o+3] };
  }

  // d_out layout (flat, f32): latA latC latP latT | recA recC recT | mA..mT | sA..sT
  float* out = (float*)d_out;
  const size_t LAT = (size_t)E * H2;   // 1,048,576
  const size_t REC = (size_t)E * DP;   // 8,388,608
  float* lat[4]; float* mm[4]; float* ss[4]; float* rec[3];
  for (int b = 0; b < 4; ++b) {
    lat[b] = out + (size_t)b * LAT;
    mm[b]  = out + 4 * LAT + 3 * REC + (size_t)b * LAT;
    ss[b]  = out + 4 * LAT + 3 * REC + 4 * LAT + (size_t)b * LAT;
  }
  for (int e = 0; e < 3; ++e) rec[e] = out + 4 * LAT + (size_t)e * REC;

  // workspace: h[E*H1] | sum[E*H2] | emb[E*256] | r1[E*H3]   (~36 MB)
  float* ws  = (float*)d_ws;
  float* h   = ws;
  float* sum = h   + (size_t)E * H1;
  float* emb = sum + (size_t)E * H2;
  float* r1  = emb + (size_t)E * 2 * H2;

  const dim3 blk(NTHREADS);

  // ---- four encoder branches ----
  for (int b = 0; b < 4; ++b) {
    // h = tanh(H @ W1^T + b1)        M=8192 N=512 K=8192
    gemm_bf16_wmma<0,0,1><<<dim3(H1 / BN, E / BM), blk, 0, stream>>>(
        H_[b], br[b].W1, br[b].b1, h, E, H1, E, E, E, H1);
    // m = h @ Wm^T + bm              M=8192 N=128 K=512
    gemm_bf16_wmma<0,0,0><<<dim3(H2 / BN, E / BM), blk, 0, stream>>>(
        h, br[b].Wm, br[b].bm, mm[b], E, H2, H1, H1, H1, H2);
    // s = h @ Ws^T + bs
    gemm_bf16_wmma<0,0,0><<<dim3(H2 / BN, E / BM), blk, 0, stream>>>(
        h, br[b].Ws, br[b].bs, ss[b], E, H2, H1, H1, H1, H2);
    // lat = tanh(eps * exp(s) + m)
    reparam_kernel<<<(E * H2 + 255) / 256, 256, 0, stream>>>(
        mm[b], ss[b], lat[b], E * H2, 0x1234567u * (b + 1));
  }

  // ---- three event branches (A, C, T) ----
  const float* Hp[3]   = { H_[0], H_[1], H_[3] };
  const float* latp[3] = { lat[0], lat[1], lat[3] };
  for (int e = 0; e < 3; ++e) {
    // sum = H^T @ lat                M=8192 N=128 K=8192 (trans-A, B is [K,N])
    gemm_bf16_wmma<1,1,0><<<dim3(H2 / BN, E / BM), blk, 0, stream>>>(
        Hp[e], latp[e], nullptr, sum, E, H2, E, E, H2, H2);
    // emb = concat(latP, sum)
    concat_kernel<<<(E * 2 * H2 + 255) / 256, 256, 0, stream>>>(lat[2], sum, emb, E * 2 * H2);
    // r1 = relu(emb @ W1e^T + b1e)   M=8192 N=256 K=256
    gemm_bf16_wmma<0,0,2><<<dim3(H3 / BN, E / BM), blk, 0, stream>>>(
        emb, ev[e].W1, ev[e].b1, r1, E, H3, 2 * H2, 2 * H2, 2 * H2, H3);
    // rec = r1 @ W2e^T + b2e         M=8192 N=1024 K=256
    gemm_bf16_wmma<0,0,0><<<dim3(DP / BN, E / BM), blk, 0, stream>>>(
        r1, ev[e].W2, ev[e].b2, rec[e], E, DP, H3, H3, H3, DP);
  }
}
